// HGRNBitMLP_14474039787742
// MI455X (gfx1250) — 8-bit matrix operand order repaired and checked by reading, compiled, not run on hardware
//
#include <hip/hip_runtime.h>


typedef __attribute__((ext_vector_type(8))) int v8i;

#define RMS_EPS 1e-8f
#define Q_EPS   1e-5f

#define WMMA_IU8(A, B, C) \
  __builtin_amdgcn_wmma_i32_16x16x64_iu8(true, (A), true, (B), (C), false, false)

static __device__ __forceinline__ v8i lds_frag_a(const char* tile, int subRow, int lane) {
  const unsigned int* p =
      (const unsigned int*)(tile + (subRow + (lane & 15)) * 80 + ((lane >> 4) << 3));
  v8i a;
  a[0] = p[0];  a[1] = p[1];
  a[2] = p[4];  a[3] = p[5];
  a[4] = p[8];  a[5] = p[9];
  a[6] = p[12]; a[7] = p[13];
  return a;
}
static __device__ __forceinline__ v8i lds_frag_b(const char* tile, int subCol, int lane) {
  const unsigned int* p =
      (const unsigned int*)(tile + (subCol + (lane & 15)) * 80 + ((lane >> 4) << 3));
  v8i b;
  b[0] = p[0]; b[1] = p[1]; b[2] = p[4];  b[3] = p[5];
  b[4] = p[8]; b[5] = p[9]; b[6] = p[12]; b[7] = p[13];
  return b;
}

static __device__ __forceinline__ void mma_step_gv(
    const char* tA, const char* tBg, const char* tBv, int wrow, int wcol, int lane,
    v8i& g00, v8i& g01, v8i& g10, v8i& g11,
    v8i& v00, v8i& v01, v8i& v10, v8i& v11) {
  v8i fa0 = lds_frag_a(tA, wrow * 32,      lane);
  v8i fa1 = lds_frag_a(tA, wrow * 32 + 16, lane);
  v8i fb0 = lds_frag_b(tBg, wcol * 32,      lane);
  v8i fb1 = lds_frag_b(tBg, wcol * 32 + 16, lane);
  g00 = WMMA_IU8(fa0, fb0, g00);
  g10 = WMMA_IU8(fa1, fb0, g10);
  g01 = WMMA_IU8(fa0, fb1, g01);
  g11 = WMMA_IU8(fa1, fb1, g11);
  v8i fv0 = lds_frag_b(tBv, wcol * 32,      lane);
  v8i fv1 = lds_frag_b(tBv, wcol * 32 + 16, lane);
  v00 = WMMA_IU8(fa0, fv0, v00);
  v10 = WMMA_IU8(fa1, fv0, v10);
  v01 = WMMA_IU8(fa0, fv1, v01);
  v11 = WMMA_IU8(fa1, fv1, v11);
}

static __device__ __forceinline__ void mma_step_1(
    const char* tA, const char* tB, int wrow, int wcol, int lane,
    v8i& c00, v8i& c01, v8i& c10, v8i& c11) {
  v8i fa0 = lds_frag_a(tA, wrow * 32,      lane);
  v8i fa1 = lds_frag_a(tA, wrow * 32 + 16, lane);
  v8i fb0 = lds_frag_b(tB, wcol * 32,      lane);
  v8i fb1 = lds_frag_b(tB, wcol * 32 + 16, lane);
  c00 = WMMA_IU8(fa0, fb0, c00);
  c10 = WMMA_IU8(fa1, fb0, c10);
  c01 = WMMA_IU8(fa0, fb1, c01);
  c11 = WMMA_IU8(fa1, fb1, c11);
}

__global__ __launch_bounds__(256) void absmean_partial(const float* __restrict__ w,
                                                       size_t n, float* __restrict__ part) {
  __shared__ float s[256];
  float acc = 0.f;
  for (size_t i = (size_t)blockIdx.x * 256u + threadIdx.x; i < n;
       i += (size_t)gridDim.x * 256u)
    acc += fabsf(w[i]);
  s[threadIdx.x] = acc; __syncthreads();
  for (int o = 128; o > 0; o >>= 1) {
    if ((int)threadIdx.x < o) s[threadIdx.x] += s[threadIdx.x + o];
    __syncthreads();
  }
  if (threadIdx.x == 0) part[blockIdx.x] = s[0];
}

__global__ __launch_bounds__(256) void reduce_final(const float* __restrict__ part,
                                                    int n, float* __restrict__ out) {
  __shared__ float s[256];
  float acc = 0.f;
  for (int i = threadIdx.x; i < n; i += 256) acc += part[i];
  s[threadIdx.x] = acc; __syncthreads();
  for (int o = 128; o > 0; o >>= 1) {
    if ((int)threadIdx.x < o) s[threadIdx.x] += s[threadIdx.x + o];
    __syncthreads();
  }
  if (threadIdx.x == 0) out[0] = s[0];
}

__global__ __launch_bounds__(256) void quant_w(const float* __restrict__ w,
                                               signed char* __restrict__ q,
                                               const float* __restrict__ wsum,
                                               float invN, size_t n) {
  const float mean = fmaxf(wsum[0] * invN, Q_EPS);
  const float s = 1.0f / mean;
  for (size_t i = (size_t)blockIdx.x * 256u + threadIdx.x; i < n;
       i += (size_t)gridDim.x * 256u) {
    float v = rintf(w[i] * s);
    v = fminf(fmaxf(v, -1.0f), 1.0f);
    q[i] = (signed char)(int)v;
  }
}

__global__ __launch_bounds__(256) void rmsq(const float* __restrict__ x,
                                            signed char* __restrict__ q,
                                            float* __restrict__ rdeq, int rowlen) {
  __shared__ float s1[256]; __shared__ float s2[256];
  const int row = blockIdx.x;
  const float* xr = x + (size_t)row * rowlen;
  float ss = 0.f, am = 0.f;
  for (int i = threadIdx.x; i < rowlen; i += 256) {
    float v = xr[i]; ss += v * v; am = fmaxf(am, fabsf(v));
  }
  s1[threadIdx.x] = ss; s2[threadIdx.x] = am; __syncthreads();
  for (int o = 128; o > 0; o >>= 1) {
    if ((int)threadIdx.x < o) {
      s1[threadIdx.x] += s1[threadIdx.x + o];
      s2[threadIdx.x] = fmaxf(s2[threadIdx.x], s2[threadIdx.x + o]);
    }
    __syncthreads();
  }
  const float rs   = rsqrtf(s1[0] / (float)rowlen + RMS_EPS);
  const float cmax = fmaxf(s2[0] * rs, Q_EPS);
  const float qs   = 127.0f / cmax;
  if (threadIdx.x == 0) rdeq[row] = cmax * (1.0f / 127.0f);
  signed char* qr = q + (size_t)row * rowlen;
  for (int i = threadIdx.x; i < rowlen; i += 256) {
    float v = rintf(xr[i] * rs * qs);
    v = fminf(fmaxf(v, -128.0f), 127.0f);
    qr[i] = (signed char)(int)v;
  }
}

__global__ __launch_bounds__(256) void gemm1_swiglu(
    const signed char* __restrict__ qx,
    const signed char* __restrict__ qwg,
    const float* __restrict__ rx,
    const float* __restrict__ wsum,
    float* __restrict__ z)
{
  constexpr int KB = 2048, KT = KB / 64, LDA = 80;
  __shared__ __align__(16) char sA[2][128 * LDA];
  __shared__ __align__(16) char sBg[2][64 * LDA];
  __shared__ __align__(16) char sBv[2][64 * LDA];

  const int tid = threadIdx.x, lane = tid & 31, wid = tid >> 5;
  const int wrow = wid >> 1, wcol = wid & 1;
  const int m0 = blockIdx.x * 128, n0 = blockIdx.y * 64;

  const int aR0 = tid >> 2, aR1 = aR0 + 64, aC = (tid & 3) << 4;
  const int bR = tid >> 2,  bC = aC;

  const char* gA  = (const char*)qx  + (size_t)m0 * KB;
  const char* gBg = (const char*)qwg + (size_t)n0 * KB;
  const char* gBv = (const char*)qwg + (size_t)(n0 + 8192) * KB;

  uint4 pA0 = *(const uint4*)(gA  + (size_t)aR0 * KB + aC);
  uint4 pA1 = *(const uint4*)(gA  + (size_t)aR1 * KB + aC);
  uint4 pBg = *(const uint4*)(gBg + (size_t)bR  * KB + bC);
  uint4 pBv = *(const uint4*)(gBv + (size_t)bR  * KB + bC);
  *(uint4*)&sA[0][aR0 * LDA + aC]  = pA0;
  *(uint4*)&sA[0][aR1 * LDA + aC]  = pA1;
  *(uint4*)&sBg[0][bR * LDA + bC]  = pBg;
  *(uint4*)&sBv[0][bR * LDA + bC]  = pBv;

  const v8i vz = {0, 0, 0, 0, 0, 0, 0, 0};
  v8i g00 = vz, g01 = vz, g10 = vz, g11 = vz;
  v8i v00 = vz, v01 = vz, v10 = vz, v11 = vz;

#pragma clang loop unroll(disable)
  for (int kt = 0; kt < KT; kt += 2) {
    __syncthreads();
    {
      const int ko = (kt + 1) * 64;
      pA0 = *(const uint4*)(gA  + (size_t)aR0 * KB + ko + aC);
      pA1 = *(const uint4*)(gA  + (size_t)aR1 * KB + ko + aC);
      pBg = *(const uint4*)(gBg + (size_t)bR  * KB + ko + bC);
      pBv = *(const uint4*)(gBv + (size_t)bR  * KB + ko + bC);
    }
    mma_step_gv(sA[0], sBg[0], sBv[0], wrow, wcol, lane,
                g00, g01, g10, g11, v00, v01, v10, v11);
    *(uint4*)&sA[1][aR0 * LDA + aC]  = pA0;
    *(uint4*)&sA[1][aR1 * LDA + aC]  = pA1;
    *(uint4*)&sBg[1][bR * LDA + bC]  = pBg;
    *(uint4*)&sBv[1][bR * LDA + bC]  = pBv;

    __syncthreads();
    const bool pf = (kt + 2 < KT);
    if (pf) {
      const int ko = (kt + 2) * 64;
      pA0 = *(const uint4*)(gA  + (size_t)aR0 * KB + ko + aC);
      pA1 = *(const uint4*)(gA  + (size_t)aR1 * KB + ko + aC);
      pBg = *(const uint4*)(gBg + (size_t)bR  * KB + ko + bC);
      pBv = *(const uint4*)(gBv + (size_t)bR  * KB + ko + bC);
    }
    mma_step_gv(sA[1], sBg[1], sBv[1], wrow, wcol, lane,
                g00, g01, g10, g11, v00, v01, v10, v11);
    if (pf) {
      *(uint4*)&sA[0][aR0 * LDA + aC]  = pA0;
      *(uint4*)&sA[0][aR1 * LDA + aC]  = pA1;
      *(uint4*)&sBg[0][bR * LDA + bC]  = pBg;
      *(uint4*)&sBv[0][bR * LDA + bC]  = pBv;
    }
  }

  const float mw = fmaxf(wsum[0] * (1.0f / 33554432.0f), Q_EPS);
  const int nl = lane & 15, hi = lane >> 4;
  v8i ag[2][2] = {{g00, g01}, {g10, g11}};
  v8i av[2][2] = {{v00, v01}, {v10, v11}};
#pragma unroll
  for (int mi = 0; mi < 2; ++mi) {
#pragma unroll
    for (int g = 0; g < 8; ++g) {
      const int m = m0 + wrow * 32 + mi * 16 + g + hi * 8;
      const float s = rx[m] * mw;
#pragma unroll
      for (int ni = 0; ni < 2; ++ni) {
        const int n = n0 + wcol * 32 + ni * 16 + nl;
        const float gv = (float)ag[mi][ni][g] * s;
        const float vv = (float)av[mi][ni][g] * s;
        const float sig = 1.0f / (1.0f + __expf(-gv));
        z[(size_t)m * 8192 + n] = gv * sig * vv;
      }
    }
  }
}

__global__ __launch_bounds__(256) void gemm2_out(
    const signed char* __restrict__ qz,
    const signed char* __restrict__ qwd,
    const float* __restrict__ rz,
    const float* __restrict__ wsum,
    float* __restrict__ out)
{
  constexpr int KB = 8192, KT = KB / 64, LDA = 80;
  __shared__ __align__(16) char sA[2][128 * LDA];
  __shared__ __align__(16) char sB[2][64 * LDA];

  const int tid = threadIdx.x, lane = tid & 31, wid = tid >> 5;
  const int wrow = wid >> 1, wcol = wid & 1;
  const int m0 = blockIdx.x * 128, n0 = blockIdx.y * 64;

  const int aR0 = tid >> 2, aR1 = aR0 + 64, aC = (tid & 3) << 4;
  const int bR = tid >> 2,  bC = aC;

  const char* gA = (const char*)qz  + (size_t)m0 * KB;
  const char* gB = (const char*)qwd + (size_t)n0 * KB;

  uint4 pA0 = *(const uint4*)(gA + (size_t)aR0 * KB + aC);
  uint4 pA1 = *(const uint4*)(gA + (size_t)aR1 * KB + aC);
  uint4 pB  = *(const uint4*)(gB + (size_t)bR  * KB + bC);
  *(uint4*)&sA[0][aR0 * LDA + aC] = pA0;
  *(uint4*)&sA[0][aR1 * LDA + aC] = pA1;
  *(uint4*)&sB[0][bR * LDA + bC]  = pB;

  const v8i vz = {0, 0, 0, 0, 0, 0, 0, 0};
  v8i c00 = vz, c01 = vz, c10 = vz, c11 = vz;

#pragma clang loop unroll(disable)
  for (int kt = 0; kt < KT; kt += 2) {
    __syncthreads();
    {
      const int ko = (kt + 1) * 64;
      pA0 = *(const uint4*)(gA + (size_t)aR0 * KB + ko + aC);
      pA1 = *(const uint4*)(gA + (size_t)aR1 * KB + ko + aC);
      pB  = *(const uint4*)(gB + (size_t)bR  * KB + ko + bC);
    }
    mma_step_1(sA[0], sB[0], wrow, wcol, lane, c00, c01, c10, c11);
    *(uint4*)&sA[1][aR0 * LDA + aC] = pA0;
    *(uint4*)&sA[1][aR1 * LDA + aC] = pA1;
    *(uint4*)&sB[1][bR * LDA + bC]  = pB;

    __syncthreads();
    const bool pf = (kt + 2 < KT);
    if (pf) {
      const int ko = (kt + 2) * 64;
      pA0 = *(const uint4*)(gA + (size_t)aR0 * KB + ko + aC);
      pA1 = *(const uint4*)(gA + (size_t)aR1 * KB + ko + aC);
      pB  = *(const uint4*)(gB + (size_t)bR  * KB + ko + bC);
    }
    mma_step_1(sA[1], sB[1], wrow, wcol, lane, c00, c01, c10, c11);
    if (pf) {
      *(uint4*)&sA[0][aR0 * LDA + aC] = pA0;
      *(uint4*)&sA[0][aR1 * LDA + aC] = pA1;
      *(uint4*)&sB[0][bR * LDA + bC]  = pB;
    }
  }

  const float mw = fmaxf(wsum[0] * (1.0f / 16777216.0f), Q_EPS);
  const int nl = lane & 15, hi = lane >> 4;
  v8i ac[2][2] = {{c00, c01}, {c10, c11}};
#pragma unroll
  for (int mi = 0; mi < 2; ++mi) {
#pragma unroll
    for (int g = 0; g < 8; ++g) {
      const int m = m0 + wrow * 32 + mi * 16 + g + hi * 8;
      const float s = rz[m] * mw;
#pragma unroll
      for (int ni = 0; ni < 2; ++ni) {
        const int n = n0 + wcol * 32 + ni * 16 + nl;
        out[(size_t)m * 2048 + n] = (float)ac[mi][ni][g] * s;
      }
    }
  }
}

extern "C" void kernel_launch(void* const* d_in, const int* in_sizes, int n_in,
                              void* d_out, int out_size, void* d_ws, size_t ws_size,
                              hipStream_t stream) {
  const float* x  = (const float*)d_in[0];
  const float* wg = (const float*)d_in[1];
  const float* wd = (const float*)d_in[2];
  float* out = (float*)d_out;

  const size_t NWG = (size_t)16384 * 2048;
  const size_t NWD = (size_t)2048 * 8192;

  char* ws = (char*)d_ws;
  size_t off = 0;
  float* red  = (float*)(ws + off); off += 256;
  float* part = (float*)(ws + off); off += 4096 * sizeof(float);
  float* rx   = (float*)(ws + off); off += 8192 * sizeof(float);
  float* rz   = (float*)(ws + off); off += 8192 * sizeof(float);
  signed char* qx  = (signed char*)(ws + off); off += (size_t)8192 * 2048;
  signed char* qwg = (signed char*)(ws + off); off += NWG;
  signed char* qwd = (signed char*)(ws + off); off += NWD;
  signed char* qz  = (signed char*)(ws + off); off += (size_t)8192 * 8192;
  float* zbuf = (float*)(ws + off); off += (size_t)8192 * 8192 * sizeof(float);

  absmean_partial<<<2048, 256, 0, stream>>>(wg, NWG, part);
  absmean_partial<<<2048, 256, 0, stream>>>(wd, NWD, part + 2048);
  reduce_final<<<1, 256, 0, stream>>>(part, 2048, red + 0);
  reduce_final<<<1, 256, 0, stream>>>(part + 2048, 2048, red + 1);

  quant_w<<<8192, 256, 0, stream>>>(wg, qwg, red + 0, 1.0f / (float)NWG, NWG);
  quant_w<<<4096, 256, 0, stream>>>(wd, qwd, red + 1, 1.0f / (float)NWD, NWD);

  rmsq<<<8192, 256, 0, stream>>>(x, qx, rx, 2048);

  gemm1_swiglu<<<dim3(64, 128), 256, 0, stream>>>(qx, qwg, rx, red + 0, zbuf);

  rmsq<<<8192, 256, 0, stream>>>(zbuf, qz, rz, 8192);

  gemm2_out<<<dim3(64, 32), 256, 0, stream>>>(qz, qwd, rz, red + 1, out);
}
